// Bidirect_LSTM_18915035971917
// MI455X (gfx1250) — matrix operand order repaired and checked by reading, not compiled, not run on hardware
//
#include <hip/hip_runtime.h>
#include <hip/hip_bf16.h>

typedef __bf16 bf16_t;
typedef __attribute__((ext_vector_type(8)))  __bf16 v8bf;
typedef __attribute__((ext_vector_type(16))) __bf16 v16bf;
typedef __attribute__((ext_vector_type(8)))  float  v8f;

static constexpr int VOCABN = 50000;
static constexpr int EMB    = 1024;
static constexpr int HID    = 2048;
static constexpr int BATCH  = 256;
static constexpr int SEQ    = 128;
static constexpr int G4H    = 4 * HID;      // 8192 weight rows (i,f,g,o)
static constexpr int KCAT   = EMB + HID;    // 3072 concat K

__device__ __forceinline__ float sigf(float x) { return 1.0f / (1.0f + __expf(-x)); }

// ---------------------------------------------------------------------------
// Embedding gather, time-reversed, fp32 -> bf16.  Xr[k][b][e] = T[idx(b,S-1-k)][e]
// ---------------------------------------------------------------------------
__global__ void gather_embed_rev(const int* __restrict__ inputs,
                                 const float* __restrict__ table,
                                 bf16_t* __restrict__ Xr)
{
    const int pair = blockIdx.x;              // SEQ*BATCH blocks
    const int k = pair / BATCH;
    const int b = pair % BATCH;
    int idx = inputs[b * SEQ + (SEQ - 1 - k)];
    if (idx > VOCABN) idx = 0;                // reference: only '> VOCAB' clamps
    const float4* src = (const float4*)(table + (size_t)idx * EMB);
    bf16_t* dst = Xr + ((size_t)k * BATCH + b) * EMB;
    for (int e4 = threadIdx.x; e4 < EMB / 4; e4 += blockDim.x) {
        float4 v = src[e4];
        dst[e4 * 4 + 0] = (bf16_t)v.x;
        dst[e4 * 4 + 1] = (bf16_t)v.y;
        dst[e4 * 4 + 2] = (bf16_t)v.z;
        dst[e4 * 4 + 3] = (bf16_t)v.w;
    }
}

// Wcat[r][0:EMB] = W_ih[r], Wcat[r][EMB:EMB+HID] = W_hh[r]   (fp32 -> bf16)
__global__ void build_wcat(const float* __restrict__ Wih,
                           const float* __restrict__ Whh,
                           bf16_t* __restrict__ Wcat)
{
    const int r = blockIdx.x;                 // G4H blocks
    const float* sih = Wih + (size_t)r * EMB;
    const float* shh = Whh + (size_t)r * HID;
    bf16_t* dst = Wcat + (size_t)r * KCAT;
    for (int c = threadIdx.x; c < EMB; c += blockDim.x) dst[c] = (bf16_t)sih[c];
    for (int c = threadIdx.x; c < HID; c += blockDim.x) dst[EMB + c] = (bf16_t)shh[c];
}

__global__ void conv_w_f(const float* __restrict__ W, bf16_t* __restrict__ Wb)
{
    const int r = blockIdx.x;                 // G4H blocks
    const float* s = W + (size_t)r * EMB;
    bf16_t* d = Wb + (size_t)r * EMB;
    for (int c = threadIdx.x; c < EMB; c += blockDim.x) d[c] = (bf16_t)s[c];
}

__global__ void add_bias(const float* __restrict__ a, const float* __restrict__ b,
                         float* __restrict__ o, int n)
{
    int i = blockIdx.x * blockDim.x + threadIdx.x;
    if (i < n) o[i] = a[i] + b[i];
}

// ---------------------------------------------------------------------------
// One LSTM step (c == 0): h = sigmoid(o) * tanh( sigmoid(i) * tanh(g) )
// gates = [x|h_prev] @ [W_ih|W_hh]^T + bias, f-gate skipped (dead: c==0).
// Per wave: 32x16 h-tile -> 2 A fragments x 3 gate B fragments = 6 accumulators.
// Grid: 128 blocks x 256 threads (8 waves) = 1024 waves = 8 Mtiles x 128 Jtiles.
// K loop is split into a branch-free X loop and a branch-free H loop; all
// pointers are bumped by 64 B per iteration so loads use immediate offsets.
// ---------------------------------------------------------------------------
struct Frags {
    const bf16_t* a0;   // A rows m0..m0+15   (lane base, +half*16 applied)
    const bf16_t* a1;   // A rows m0+16..m0+31
    const bf16_t* wi;   // gate-i weight row (lane base, +half*16 applied)
    const bf16_t* wg;   // gate-g
    const bf16_t* wo;   // gate-o
};

__device__ __forceinline__ void wmma6(Frags& f,
                                      v8f& ai0, v8f& ai1,
                                      v8f& ag0, v8f& ag1,
                                      v8f& ao0, v8f& ao1)
{
    union { v16bf v; v8bf h[2]; } a0, a1;
    // A contiguous K: lane<16 K {0..15}; lane>=16 K {16..31}
    a0.h[0] = *(const v8bf*)(f.a0);
    a0.h[1] = *(const v8bf*)(f.a0 + 8);
    a1.h[0] = *(const v8bf*)(f.a1);
    a1.h[1] = *(const v8bf*)(f.a1 + 8);
    // B: lane<16 -> col N=l15, K contiguous from row-major W
    v16bf bi = *(const v16bf*)(f.wi);
    v16bf bg = *(const v16bf*)(f.wg);
    v16bf bo = *(const v16bf*)(f.wo);

    ai0 = __builtin_amdgcn_wmma_f32_16x16x32_bf16(false, a0.v, false, bi, (short)0, ai0, false, false);
    ai1 = __builtin_amdgcn_wmma_f32_16x16x32_bf16(false, a1.v, false, bi, (short)0, ai1, false, false);
    ag0 = __builtin_amdgcn_wmma_f32_16x16x32_bf16(false, a0.v, false, bg, (short)0, ag0, false, false);
    ag1 = __builtin_amdgcn_wmma_f32_16x16x32_bf16(false, a1.v, false, bg, (short)0, ag1, false, false);
    ao0 = __builtin_amdgcn_wmma_f32_16x16x32_bf16(false, a0.v, false, bo, (short)0, ao0, false, false);
    ao1 = __builtin_amdgcn_wmma_f32_16x16x32_bf16(false, a1.v, false, bo, (short)0, ao1, false, false);

    // bump one K-tile (32 bf16) forward
    f.a0 += 32; f.a1 += 32; f.wi += 32; f.wg += 32; f.wo += 32;
}

__global__ __launch_bounds__(256)
void lstm_wmma_step(const bf16_t* __restrict__ X,    // [BATCH, EMB]
                    const bf16_t* __restrict__ H,    // [BATCH, HID] or nullptr
                    const bf16_t* __restrict__ W,    // [G4H, ldw] row-major bf16
                    int ldw,
                    const float* __restrict__ bias,  // [G4H]
                    int Kh,                          // 0 or HID
                    bf16_t* __restrict__ Hout,       // [BATCH, HID] or nullptr
                    float* __restrict__ Fout,        // [BATCH, 2*HID] or nullptr
                    int fout_off)
{
    const int lane = threadIdx.x & 31;
    const int half = lane >> 4;
    const int l15  = lane & 15;
    const int gw   = blockIdx.x * 8 + (threadIdx.x >> 5);  // 0..1023
    const int m0   = (gw & 7) * 32;                        // 8 M-tiles of 32 rows
    const int j0   = (gw >> 3) * 16;                       // 128 J-tiles (h columns)

    v8f ai0 = {}, ag0 = {}, ao0 = {};   // M rows m0..m0+15
    v8f ai1 = {}, ag1 = {}, ao1 = {};   // M rows m0+16..m0+31

    Frags f;
    f.wi = W + (size_t)(0 * HID + j0 + l15) * ldw + half * 16;
    f.wg = W + (size_t)(2 * HID + j0 + l15) * ldw + half * 16;
    f.wo = W + (size_t)(3 * HID + j0 + l15) * ldw + half * 16;
    f.a0 = X + (size_t)(m0 + l15) * EMB + half * 16;
    f.a1 = X + (size_t)(m0 + 16 + l15) * EMB + half * 16;

    // ---- K over x: EMB/32 = 32 branch-free iterations ----
    #pragma unroll 4
    for (int kt = 0; kt < EMB / 32; ++kt)
        wmma6(f, ai0, ai1, ag0, ag1, ao0, ao1);

    // ---- K over h_prev: Kh/32 branch-free iterations (0 when h==0) ----
    if (Kh > 0) {
        f.a0 = H + (size_t)(m0 + l15) * HID + half * 16;
        f.a1 = H + (size_t)(m0 + 16 + l15) * HID + half * 16;
        #pragma unroll 4
        for (int kt = 0; kt < Kh / 32; ++kt)
            wmma6(f, ai0, ai1, ag0, ag1, ao0, ao1);
    }

    const int   col = j0 + l15;
    const float bii = bias[col];
    const float big = bias[2 * HID + col];
    const float bio = bias[3 * HID + col];

    // D layout: lane<16 -> N=l15, VGPR v -> M = v (+8 for lanes>=16)
    const int r0 = m0 + half * 8;            // rows of block 0
    bf16_t* hp0 = Hout ? Hout + (size_t)r0 * HID + col : nullptr;
    bf16_t* hp1 = Hout ? Hout + (size_t)(r0 + 16) * HID + col : nullptr;
    float*  fp0 = Fout ? Fout + (size_t)r0 * (2 * HID) + fout_off + col : nullptr;
    float*  fp1 = Fout ? Fout + (size_t)(r0 + 16) * (2 * HID) + fout_off + col : nullptr;

    #pragma unroll
    for (int v = 0; v < 8; ++v) {
        float c0 = sigf(ai0[v] + bii) * tanhf(ag0[v] + big);
        float h0 = sigf(ao0[v] + bio) * tanhf(c0);
        float c1 = sigf(ai1[v] + bii) * tanhf(ag1[v] + big);
        float h1 = sigf(ao1[v] + bio) * tanhf(c1);
        if (Hout) {
            hp0[(size_t)v * HID] = (bf16_t)h0;
            hp1[(size_t)v * HID] = (bf16_t)h1;
        }
        if (Fout) {
            fp0[(size_t)v * (2 * HID)] = h0;
            fp1[(size_t)v * (2 * HID)] = h1;
        }
    }
}

// ---------------------------------------------------------------------------
extern "C" void kernel_launch(void* const* d_in, const int* in_sizes, int n_in,
                              void* d_out, int out_size, void* d_ws, size_t ws_size,
                              hipStream_t stream)
{
    (void)in_sizes; (void)n_in; (void)out_size; (void)ws_size;
    const int*   inputs = (const int*)d_in[0];
    const float* table  = (const float*)d_in[1];
    const float* Wihf   = (const float*)d_in[2];
    // d_in[3] = W_hh_f: multiplied by h==0, never needed
    const float* bihf   = (const float*)d_in[4];
    const float* bhhf   = (const float*)d_in[5];
    const float* Wihb   = (const float*)d_in[6];
    const float* Whhb   = (const float*)d_in[7];
    const float* bihb   = (const float*)d_in[8];
    const float* bhhb   = (const float*)d_in[9];
    float* out = (float*)d_out;

    // workspace carve-up (~131 MB)
    char* ws = (char*)d_ws;
    size_t off = 0;
    auto carve = [&](size_t bytes) -> void* {
        void* p = ws + off;
        off = (off + bytes + 255) & ~(size_t)255;
        return p;
    };
    bf16_t* Xr   = (bf16_t*)carve((size_t)SEQ * BATCH * EMB * sizeof(bf16_t)); // 64 MB
    bf16_t* Wcat = (bf16_t*)carve((size_t)G4H * KCAT * sizeof(bf16_t));        // 48 MB
    bf16_t* Wf   = (bf16_t*)carve((size_t)G4H * EMB * sizeof(bf16_t));         // 16 MB
    float*  bb   = (float*)carve((size_t)G4H * sizeof(float));
    float*  bfv  = (float*)carve((size_t)G4H * sizeof(float));
    bf16_t* H0   = (bf16_t*)carve((size_t)BATCH * HID * sizeof(bf16_t));       // 1 MB
    bf16_t* H1   = (bf16_t*)carve((size_t)BATCH * HID * sizeof(bf16_t));       // 1 MB

    // preprocessing (one-time per call; weights then live in L2)
    gather_embed_rev<<<SEQ * BATCH, 256, 0, stream>>>(inputs, table, Xr);
    build_wcat<<<G4H, 256, 0, stream>>>(Wihb, Whhb, Wcat);
    conv_w_f<<<G4H, 256, 0, stream>>>(Wihf, Wf);
    add_bias<<<(G4H + 255) / 256, 256, 0, stream>>>(bihb, bhhb, bb, G4H);
    add_bias<<<(G4H + 255) / 256, 256, 0, stream>>>(bihf, bhhf, bfv, G4H);

    // forward cell: x = embeds[:, SEQ-1] = Xr[0], h = c = 0 -> out[:, 0:HID]
    lstm_wmma_step<<<128, 256, 0, stream>>>(Xr, nullptr, Wf, EMB, bfv,
                                            /*Kh=*/0, nullptr, out, /*off=*/0);

    // backward recurrence over reversed sequence; c stays 0 every step
    bf16_t* hb[2] = {H0, H1};
    for (int k = 0; k < SEQ; ++k) {
        const bf16_t* Xk   = Xr + (size_t)k * BATCH * EMB;
        const bf16_t* Hin  = (k == 0) ? nullptr : hb[(k - 1) & 1];
        const int     Kh   = (k == 0) ? 0 : HID;           // h0 == 0: skip h GEMM
        bf16_t*       Hnew = hb[k & 1];
        float*        Fo   = (k == SEQ - 1) ? out : nullptr;
        lstm_wmma_step<<<128, 256, 0, stream>>>(Xk, Hin, Wcat, KCAT, bb,
                                                Kh, Hnew, Fo, /*off=*/HID);
    }
}
